// PromptWindowAttention_25340307046823
// MI455X (gfx1250) — matrix operand order repaired and checked by reading, not compiled, not run on hardware
//
#include <hip/hip_runtime.h>
#include <hip/hip_bf16.h>

// ---------------------------------------------------------------------------
// PromptWindowAttention fused kernel for gfx1250 (MI455X, CDNA5, wave32).
// One workgroup (8 waves, 256 threads) handles one window (batch element).
// All intermediates live in LDS; matmuls use v_wmma_f32_16x16x32_bf16.
// V is stored transposed (head-dim-major) so all WMMA fragments load as
// aligned ds_load_b128 pairs -- no scalar LDS gathers.
// ---------------------------------------------------------------------------

#define NTOK   54      // tokens per window (49 + 5 prompts)
#define NPAD   64      // padded token count (4 WMMA M-tiles)
#define DIM    384
#define NH     12
#define HD     32
#define NPROMPT 5
#define NWIN   64      // number of shifted-window masks
#define XS_STRIDE 392  // bf16 elems/row (384 + pad); 784B = 49*16 (16B-aligned rows)
#define QS_STRIDE 776  // bf16 elems/row for Q|K (768 + pad); 1552B = 97*16
#define VT_STRIDE 64   // vT: 384 rows (head-dim) x 64 tokens
#define ATTN_SCALE 0.17677669529663687f  // 32^-0.5

typedef __attribute__((ext_vector_type(16))) __bf16 v16bf;
typedef __attribute__((ext_vector_type(8)))  __bf16 v8bf;
typedef __attribute__((ext_vector_type(8)))  float  v8f;
typedef __attribute__((ext_vector_type(4)))  float  v4f;

union V16U { v16bf v; v8bf h[2]; };

__device__ __forceinline__ v8f zero8() {
    v8f c;
#pragma unroll
    for (int i = 0; i < 8; ++i) c[i] = 0.0f;
    return c;
}

__device__ __forceinline__ v8f wmma_bf16(v16bf a, v16bf b, v8f c) {
    // 8 args: (neg_a, A, neg_b, B, c_mod, C, reuse_a, reuse_b)
    return __builtin_amdgcn_wmma_f32_16x16x32_bf16(false, a, false, b, (short)0, c,
                                                   false, false);
}

// A fragment (16x32 bf16, rows M, reduction K) from row-major bf16 (stride elems).
// ISA layout: lanes 0-15 row M=lane, K chunks [8g..8g+7] and [16+8g..16+8g+7], g=lane>>4.
__device__ __forceinline__ v16bf load_a_frag(const __bf16* base, int stride,
                                             int m0, int k0, int lane) {
    int m = m0 + (lane & 15);
    int g = lane >> 4;
    const __bf16* p = base + m * stride + k0 + 8 * g;
    V16U r;
    r.h[0] = *(const v8bf*)(p);
    r.h[1] = *(const v8bf*)(p + 16);
    return r.v;
}

// B fragment (32x16: K x N) where B(k,n) = src[n*stride + k] (rows n, contiguous k).
// ISA layout: lane holds N = lane&15, K chunks [8g..8g+7] and [16+8g..16+8g+7], g=lane>>4.
__device__ __forceinline__ v16bf load_bT_frag(const __bf16* base, int stride,
                                              int n0, int k0, int lane) {
    int n = n0 + (lane & 15);
    int g = lane >> 4;
    const __bf16* p = base + n * stride + k0 + 8 * g;
    V16U r;
    r.h[0] = *(const v8bf*)(p);
    r.h[1] = *(const v8bf*)(p + 16);
    return r.v;
}

// Same as load_bT_frag (K chunks [8g..8g+7], [16+8g..16+8g+7], same as load_a_frag) but source is f32 global weights W[n][k]; convert to bf16.
__device__ __forceinline__ v16bf load_bT_frag_f32(const float* base, int stride,
                                                  int n0, int k0, int lane) {
    int n = n0 + (lane & 15);
    int g = lane >> 4;
    const float* p = base + n * stride + k0 + 8 * g;
    v4f f0 = *(const v4f*)(p);
    v4f f1 = *(const v4f*)(p + 4);
    v4f f2 = *(const v4f*)(p + 16);
    v4f f3 = *(const v4f*)(p + 20);
    v16bf r;
#pragma unroll
    for (int i = 0; i < 4; ++i) {
        r[i]      = (__bf16)f0[i];
        r[4 + i]  = (__bf16)f1[i];
        r[8 + i]  = (__bf16)f2[i];
        r[12 + i] = (__bf16)f3[i];
    }
    return r;
}

__global__ __launch_bounds__(256)
void pwa_fused_kernel(const float* __restrict__ x,
                      const float* __restrict__ mask,
                      const float* __restrict__ qkv_w,
                      const float* __restrict__ qkv_b,
                      const float* __restrict__ proj_w,
                      const float* __restrict__ proj_b,
                      const float* __restrict__ rpb,
                      float* __restrict__ out) {
    extern __shared__ char smem[];
    // [0,      50176): xs = x as bf16, 64 x XS_STRIDE. Reused as attn output later.
    // [50176, 149504): qs = Q|K as bf16, 64 x QS_STRIDE (cols 0..383 Q, 384..767 K).
    // [149504,198656): vT = V transposed, 384 (head-dim rows) x 64 (tokens).
    // [198656,215040): per-wave softmax-P buffers, 8 x (16 x 64) bf16.
    __bf16* xs   = (__bf16*)(smem);
    __bf16* qs   = (__bf16*)(smem + 50176);
    __bf16* vT   = (__bf16*)(smem + 149504);
    __bf16* pall = (__bf16*)(smem + 198656);

    const int b    = blockIdx.x;
    const int tid  = threadIdx.x;
    const int wave = tid >> 5;
    const int lane = tid & 31;
    const int g    = lane >> 4;
    const int l16  = lane & 15;

    // ---------------- Phase 0: load x -> LDS bf16, zero pad rows ----------------
    {
        const float* xb = x + (size_t)b * NTOK * DIM;
        for (int i = tid; i < (NTOK * DIM) / 4; i += 256) {
            v4f f = *(const v4f*)(xb + i * 4);
            int row = (i * 4) / DIM;
            int col = (i * 4) % DIM;
            __bf16* d = xs + row * XS_STRIDE + col;
            d[0] = (__bf16)f[0];
            d[1] = (__bf16)f[1];
            d[2] = (__bf16)f[2];
            d[3] = (__bf16)f[3];
        }
        for (int i = tid; i < (NPAD - NTOK) * DIM; i += 256) {
            int row = NTOK + i / DIM;
            int col = i % DIM;
            xs[row * XS_STRIDE + col] = (__bf16)0.0f;
        }
    }
    __syncthreads();

    // ---------------- Phase 1: QKV GEMM (64 x 384) x (384 x 1152) --------------
    // 4 M-tiles x 72 N-tiles = 288 tiles over 8 waves.
    // N-tiles 0..47 (Q,K) land row-major in qs; 48..71 (V) land transposed in vT.
    for (int t = wave; t < 4 * 72; t += 8) {
        int mt = t / 72;
        int nt = t % 72;
        v8f acc = zero8();
#pragma unroll
        for (int kt = 0; kt < 12; ++kt) {
            v16bf a  = load_a_frag(xs, XS_STRIDE, mt * 16, kt * 32, lane);
            v16bf bf = load_bT_frag_f32(qkv_w, DIM, nt * 16, kt * 32, lane);
            acc = wmma_bf16(a, bf, acc);
        }
        int n_col = nt * 16 + l16;
        float bias = qkv_b[n_col];
        if (nt < 48) {
            // Q or K: row-major scatter (8 x ds_store_b16).
#pragma unroll
            for (int v = 0; v < 8; ++v) {
                int m = mt * 16 + v + 8 * g;
                qs[m * QS_STRIDE + n_col] = (__bf16)(acc[v] + bias);
            }
        } else {
            // V: transposed store. D-tile rows m = mt*16 + 8g + v are contiguous
            // in v -> one packed 16B ds_store_b128 per lane.
            int d = n_col - 2 * DIM;   // head-dim-major row of vT, 0..383
            v8bf pk;
#pragma unroll
            for (int v = 0; v < 8; ++v) pk[v] = (__bf16)(acc[v] + bias);
            *(v8bf*)(vT + d * VT_STRIDE + mt * 16 + 8 * g) = pk;
        }
    }
    __syncthreads();

    // ---------------- Phase 2: attention, per (head, 16-query-row) unit --------
    // 12 heads x 4 M-tiles = 48 units over 8 waves. xs region becomes attn output.
    const float* maskw = mask + (size_t)(b & (NWIN - 1)) * 49 * 49;
    __bf16* pw = pall + wave * 16 * 64;

    for (int u = wave; u < 48; u += 8) {
        int h  = u >> 2;
        int mt = u & 3;
        int qcol = h * HD;
        int kcol = DIM + h * HD;

        // S = q . k^T  (16 x 64), K = 32 -> one WMMA per 16-key tile.
        float s[4][8];
        {
            v16bf a = load_a_frag(qs, QS_STRIDE, mt * 16, qcol, lane);
#pragma unroll
            for (int nt = 0; nt < 4; ++nt) {
                v16bf bb = load_bT_frag(qs, QS_STRIDE, nt * 16, kcol, lane);
                v8f c = zero8();
                c = wmma_bf16(a, bb, c);
#pragma unroll
                for (int v = 0; v < 8; ++v) s[nt][v] = c[v];
            }
        }

        // scale + relative position bias + window mask, then row softmax.
#pragma unroll
        for (int v = 0; v < 8; ++v) {
            int iq = mt * 16 + v + 8 * g;   // query token index
            float rowmax = -1e30f;
#pragma unroll
            for (int nt = 0; nt < 4; ++nt) {
                int jk = nt * 16 + l16;     // key token index
                float val;
                if (jk >= NTOK) {
                    val = -1e30f;           // padded keys excluded
                } else {
                    val = s[nt][v] * ATTN_SCALE;
                    if (iq >= NPROMPT && iq < NTOK && jk >= NPROMPT) {
                        int iqq = iq - NPROMPT, jkk = jk - NPROMPT;
                        int r1 = iqq / 7, c1 = iqq % 7;
                        int r2 = jkk / 7, c2 = jkk % 7;
                        int idx = (r1 - r2 + 6) * 13 + (c1 - c2 + 6);
                        val += rpb[idx * NH + h];
                        val += maskw[iqq * 49 + jkk];
                    }
                }
                s[nt][v] = val;
                rowmax = fmaxf(rowmax, val);
            }
            // Row lives across the 16 lanes of this half-wave: butterfly reduce.
#pragma unroll
            for (int mdist = 1; mdist < 16; mdist <<= 1)
                rowmax = fmaxf(rowmax, __shfl_xor(rowmax, mdist, 32));
            float rowsum = 0.0f;
#pragma unroll
            for (int nt = 0; nt < 4; ++nt) {
                float e = __expf(s[nt][v] - rowmax);
                s[nt][v] = e;
                rowsum += e;
            }
#pragma unroll
            for (int mdist = 1; mdist < 16; mdist <<= 1)
                rowsum += __shfl_xor(rowsum, mdist, 32);
            float inv = 1.0f / rowsum;
            int lr = v + 8 * g;             // local row 0..15
#pragma unroll
            for (int nt = 0; nt < 4; ++nt)
                pw[lr * 64 + nt * 16 + l16] = (__bf16)(s[nt][v] * inv);
        }

        // O = P(16x64) . V(64x32). V fragments come from vT (head-dim rows,
        // token-contiguous) -> plain aligned ds_load_b128 pairs.
        v8f o0 = zero8(), o1 = zero8();
#pragma unroll
        for (int kt = 0; kt < 2; ++kt) {
            v16bf pa = load_a_frag(pw, 64, 0, kt * 32, lane);
            v16bf b0 = load_bT_frag(vT, VT_STRIDE, h * HD,      kt * 32, lane);
            v16bf b1 = load_bT_frag(vT, VT_STRIDE, h * HD + 16, kt * 32, lane);
            o0 = wmma_bf16(pa, b0, o0);
            o1 = wmma_bf16(pa, b1, o1);
        }
        // Store per-head output into reused xs region: ao[m][h*32 + d].
#pragma unroll
        for (int v = 0; v < 8; ++v) {
            int m = mt * 16 + v + 8 * g;
            xs[m * XS_STRIDE + h * HD + l16]      = (__bf16)o0[v];
            xs[m * XS_STRIDE + h * HD + 16 + l16] = (__bf16)o1[v];
        }
    }
    __syncthreads();

    // ---------------- Phase 3: output projection (64 x 384) x (384 x 384) ------
    // 4 M-tiles x 24 N-tiles = 96 tiles over 8 waves.
    float* ob = out + (size_t)b * NTOK * DIM;
    for (int t = wave; t < 4 * 24; t += 8) {
        int mt = t / 24;
        int nt = t % 24;
        v8f acc = zero8();
#pragma unroll
        for (int kt = 0; kt < 12; ++kt) {
            v16bf a  = load_a_frag(xs, XS_STRIDE, mt * 16, kt * 32, lane);
            v16bf bf = load_bT_frag_f32(proj_w, DIM, nt * 16, kt * 32, lane);
            acc = wmma_bf16(a, bf, acc);
        }
        int n_col = nt * 16 + l16;
        float bias = proj_b[n_col];
#pragma unroll
        for (int v = 0; v < 8; ++v) {
            int m = mt * 16 + v + 8 * g;
            if (m < NTOK) ob[m * DIM + n_col] = acc[v] + bias;
        }
    }
}

extern "C" void kernel_launch(void* const* d_in, const int* in_sizes, int n_in,
                              void* d_out, int out_size, void* d_ws, size_t ws_size,
                              hipStream_t stream) {
    (void)in_sizes; (void)n_in; (void)out_size; (void)d_ws; (void)ws_size;
    const float* x      = (const float*)d_in[0];
    const float* mask   = (const float*)d_in[1];
    const float* qkv_w  = (const float*)d_in[2];
    const float* qkv_b  = (const float*)d_in[3];
    const float* proj_w = (const float*)d_in[4];
    const float* proj_b = (const float*)d_in[5];
    const float* rpb    = (const float*)d_in[6];
    float* out = (float*)d_out;

    const size_t smem_bytes = 215040;  // 210 KB < 320 KB/WGP
    hipFuncSetAttribute((const void*)pwa_fused_kernel,
                        hipFuncAttributeMaxDynamicSharedMemorySize,
                        (int)smem_bytes);
    pwa_fused_kernel<<<1024, 256, smem_bytes, stream>>>(
        x, mask, qkv_w, qkv_b, proj_w, proj_b, rpb, out);
}
